// DeepSeekClone_11433202942604
// MI455X (gfx1250) — matrix operand order repaired and checked by reading, not compiled, not run on hardware
//
#include <hip/hip_runtime.h>
#include <hip/hip_bf16.h>

typedef __attribute__((ext_vector_type(16))) _Float16 v16h;
typedef __attribute__((ext_vector_type(8)))  _Float16 v8h;
typedef __attribute__((ext_vector_type(8)))  float    v8f;

#define TB_S   1024
#define TB_D   2048
#define TB_H   16
#define TB_DH  128
#define TB_E   8
#define TB_DFF 8192
#define TB_V   32000
#define TB_L   2

__device__ __forceinline__ float gelu_tanh(float x) {
    const float c0 = 0.7978845608028654f;   // sqrt(2/pi)
    float x3 = x * x * x;
    return 0.5f * x * (1.0f + tanhf(c0 * (x + 0.044715f * x3)));
}

// ---------------------------------------------------------------------------
// Generic WMMA f16 GEMM: C = epilogue(alpha * (A @ W) + bias)
//   A: [M,K] f32, element (m,k) at A[z*a_z + m*lda + k]
//   W: [K,N] f32, element (k,n) at W[z*w_z + k*ws_k + n*ws_n]
//   C: [M,N] f32, element (m,n) at C[z*c_z + m*ldc + n]
// accum==1: C += rowscale[z*rs_z + m*rs_row] * v  (atomic)
// Block: 128 threads (4 waves). Tile: 128x64x32; each wave owns a 32x64 strip
// (8 accumulators -> 8 v_wmma per k-step, B fragments reused 2x).
// Requires M%128==0, N%64==0, K%32==0.
// ---------------------------------------------------------------------------
__global__ __launch_bounds__(128)
void wmma_gemm(const float* __restrict__ A, long lda, long a_z,
               const float* __restrict__ W, long ws_k, long ws_n, long w_z,
               const float* __restrict__ bias, long bias_z,
               float* __restrict__ C, long ldc, long c_z,
               const float* __restrict__ rowscale, long rs_row, long rs_z,
               int M, int N, int Kd, float alpha, int act, int accum) {
    __shared__ _Float16 sA[128 * 40];  // [row][k], pad 32->40 halves
    __shared__ _Float16 sW[64 * 40];   // [col][k] (transposed), pad 40

    const int tid  = threadIdx.x;
    const int wv   = tid >> 5;
    const int lane = tid & 31;
    const int r    = lane & 15;
    const int sel  = lane >> 4;
    const int z    = blockIdx.z;

    const float* Az = A + (long)z * a_z;
    const float* Wz = W + (long)z * w_z;
    float*       Cz = C + (long)z * c_z;

    const int bm = blockIdx.y * 128;
    const int bn = blockIdx.x * 64;

    v8f acc[2][4] = {};

    for (int k0 = 0; k0 < Kd; k0 += 32) {
        // ---- prefetch next k-tile toward cache while this one computes ----
        if (k0 + 32 < Kd) {
            __builtin_prefetch(Az + (long)(bm + (tid >> 3) * 4) * lda + (k0 + 32), 0, 1);
            if (ws_n == 1)
                __builtin_prefetch(Wz + (long)(k0 + 32 + (tid >> 4)) * ws_k + bn, 0, 1);
            else
                __builtin_prefetch(Wz + (long)(bn + (tid >> 3)) * ws_n + (k0 + 32), 0, 1);
        }
        // ---- stage A tile (128 rows x 32 k) as f16 into LDS ----
        #pragma unroll
        for (int i = 0; i < 8; ++i) {
            int q  = tid + i * 128;        // 0..1023 quads
            int ar = q >> 3;               // row in tile 0..127
            int kq = (q & 7) << 2;         // k quad start
            float4 v = *(const float4*)(Az + (long)(bm + ar) * lda + (k0 + kq));
            _Float16* dst = &sA[ar * 40 + kq];
            dst[0] = (_Float16)v.x; dst[1] = (_Float16)v.y;
            dst[2] = (_Float16)v.z; dst[3] = (_Float16)v.w;
        }
        // ---- stage W tile (32 k x 64 n), stored transposed [n][k] ----
        if (ws_n == 1) {
            #pragma unroll
            for (int i = 0; i < 4; ++i) {
                int q  = tid + i * 128;
                int kk = q >> 4;           // 0..31
                int nq = (q & 15) << 2;    // 0..60
                float4 v = *(const float4*)(Wz + (long)(k0 + kk) * ws_k + (bn + nq));
                sW[(nq + 0) * 40 + kk] = (_Float16)v.x;
                sW[(nq + 1) * 40 + kk] = (_Float16)v.y;
                sW[(nq + 2) * 40 + kk] = (_Float16)v.z;
                sW[(nq + 3) * 40 + kk] = (_Float16)v.w;
            }
        } else {   // ws_k == 1 (W accessed transposed: contiguous along k)
            #pragma unroll
            for (int i = 0; i < 4; ++i) {
                int q  = tid + i * 128;
                int nn = q >> 3;           // 0..63
                int kq = (q & 7) << 2;
                float4 v = *(const float4*)(Wz + (long)(bn + nn) * ws_n + (k0 + kq));
                _Float16* dst = &sW[nn * 40 + kq];
                dst[0] = (_Float16)v.x; dst[1] = (_Float16)v.y;
                dst[2] = (_Float16)v.z; dst[3] = (_Float16)v.w;
            }
        }
        __syncthreads();

        // ---- build fragments and issue WMMAs ----
        // A frag: lanes 0-15: row=r, K {0..7,16..23}; lanes 16-31: row=r, K {8..15,24..31}
        v16h afrag[2];
        #pragma unroll
        for (int mt = 0; mt < 2; ++mt) {
            const int arow = wv * 32 + mt * 16 + r;
            v8h alo = *(const v8h*)&sA[arow * 40 + sel * 8];
            v8h ahi = *(const v8h*)&sA[arow * 40 + sel * 8 + 16];
            #pragma unroll
            for (int i = 0; i < 8; ++i) { afrag[mt][i] = alo[i]; afrag[mt][i + 8] = ahi[i]; }
        }

        #pragma unroll
        for (int nt = 0; nt < 4; ++nt) {
            // B frag: lanes 0-15: col=r, K {0..7,16..23}; lanes 16-31: col=r, K {8..15,24..31}
            const int c = nt * 16 + r;
            v8h blo = *(const v8h*)&sW[c * 40 + sel * 8];
            v8h bhi = *(const v8h*)&sW[c * 40 + sel * 8 + 16];
            v16h bfrag;
            #pragma unroll
            for (int i = 0; i < 8; ++i) { bfrag[i] = blo[i]; bfrag[i + 8] = bhi[i]; }
            #pragma unroll
            for (int mt = 0; mt < 2; ++mt) {
                acc[mt][nt] = __builtin_amdgcn_wmma_f32_16x16x32_f16(
                    false, afrag[mt], false, bfrag, (short)0, acc[mt][nt], false, false);
            }
        }
        __syncthreads();
    }

    // ---- epilogue ----
    #pragma unroll
    for (int nt = 0; nt < 4; ++nt) {
        const int col = bn + nt * 16 + r;
        float bv = bias ? bias[(long)z * bias_z + col] : 0.0f;
        #pragma unroll
        for (int mt = 0; mt < 2; ++mt) {
            #pragma unroll
            for (int g = 0; g < 8; ++g) {
                const int row = bm + wv * 32 + mt * 16 + sel * 8 + g;
                if (row < M && col < N) {
                    float v = acc[mt][nt][g] * alpha + bv;
                    if (act) v = gelu_tanh(v);
                    long idx = (long)row * ldc + col;
                    if (accum) {
                        float s = rowscale[(long)z * rs_z + (long)row * rs_row];
                        atomicAdd(&Cz[idx], s * v);
                    } else {
                        Cz[idx] = v;
                    }
                }
            }
        }
    }
}

// ---------------------------------------------------------------------------
__global__ __launch_bounds__(256)
void embed_kernel(const int* __restrict__ toks, const float* __restrict__ emb,
                  float* __restrict__ x) {
    int s = blockIdx.x;
    int t = toks[s];
    const float4* src = (const float4*)(emb + (long)t * TB_D);
    float4*       dst = (float4*)(x + (long)s * TB_D);
    for (int i = threadIdx.x; i < TB_D / 4; i += blockDim.x) dst[i] = src[i];
}

__global__ __launch_bounds__(256)
void ln_kernel(const float* __restrict__ x, const float* __restrict__ sc,
               const float* __restrict__ bi, float* __restrict__ y) {
    const int row = blockIdx.x, tid = threadIdx.x;
    const float* xr = x + (long)row * TB_D;
    float v[8], s = 0.f, ss = 0.f;
    #pragma unroll
    for (int i = 0; i < 8; ++i) {
        v[i] = xr[tid + i * 256];
        s += v[i]; ss += v[i] * v[i];
    }
    __shared__ float sh1[256], sh2[256];
    sh1[tid] = s; sh2[tid] = ss; __syncthreads();
    for (int st = 128; st > 0; st >>= 1) {
        if (tid < st) { sh1[tid] += sh1[tid + st]; sh2[tid] += sh2[tid + st]; }
        __syncthreads();
    }
    float mean = sh1[0] * (1.0f / TB_D);
    float var  = sh2[0] * (1.0f / TB_D) - mean * mean;
    float rs   = rsqrtf(var + 1e-6f);
    float* yr = y + (long)row * TB_D;
    #pragma unroll
    for (int i = 0; i < 8; ++i) {
        int c = tid + i * 256;
        yr[c] = (v[i] - mean) * rs * sc[c] + bi[c];
    }
}

__global__ __launch_bounds__(256)
void softmax1024(float* __restrict__ x) {
    const long row = blockIdx.x;
    float* p = x + row * 1024;
    const int tid = threadIdx.x;
    float v[4];
    #pragma unroll
    for (int i = 0; i < 4; ++i) v[i] = p[tid + i * 256];
    float m = fmaxf(fmaxf(v[0], v[1]), fmaxf(v[2], v[3]));
    __shared__ float sh[256];
    sh[tid] = m; __syncthreads();
    for (int st = 128; st > 0; st >>= 1) {
        if (tid < st) sh[tid] = fmaxf(sh[tid], sh[tid + st]);
        __syncthreads();
    }
    float rowmax = sh[0]; __syncthreads();
    float s = 0.f;
    #pragma unroll
    for (int i = 0; i < 4; ++i) { v[i] = expf(v[i] - rowmax); s += v[i]; }
    sh[tid] = s; __syncthreads();
    for (int st = 128; st > 0; st >>= 1) {
        if (tid < st) sh[tid] += sh[tid + st];
        __syncthreads();
    }
    float inv = 1.0f / sh[0];
    #pragma unroll
    for (int i = 0; i < 4; ++i) p[tid + i * 256] = v[i] * inv;
}

__global__ __launch_bounds__(256)
void gate_kernel(const float* __restrict__ xln, const float* __restrict__ gw,
                 const float* __restrict__ gb, float* __restrict__ gate) {
    const int row = blockIdx.x, tid = threadIdx.x;
    float p[TB_E];
    #pragma unroll
    for (int e = 0; e < TB_E; ++e) p[e] = 0.f;
    for (int d = tid; d < TB_D; d += 256) {
        float xv = xln[(long)row * TB_D + d];
        #pragma unroll
        for (int e = 0; e < TB_E; ++e) p[e] += xv * gw[(long)d * TB_E + e];
    }
    __shared__ float sh[256];
    for (int e = 0; e < TB_E; ++e) {
        sh[tid] = p[e]; __syncthreads();
        for (int st = 128; st > 0; st >>= 1) {
            if (tid < st) sh[tid] += sh[tid + st];
            __syncthreads();
        }
        if (tid == 0) gate[(long)row * TB_E + e] = sh[0] + gb[e];
        __syncthreads();
    }
}

__global__ __launch_bounds__(256)
void moe_topk_kernel(const float* __restrict__ gate, float* __restrict__ ew) {
    int t = blockIdx.x * blockDim.x + threadIdx.x;
    if (t >= TB_S) return;
    float g[TB_E];
    float mx = -1e30f;
    #pragma unroll
    for (int e = 0; e < TB_E; ++e) { g[e] = gate[(long)t * TB_E + e]; mx = fmaxf(mx, g[e]); }
    float s = 0.f;
    #pragma unroll
    for (int e = 0; e < TB_E; ++e) { g[e] = expf(g[e] - mx); s += g[e]; }
    float inv = 1.0f / s;
    #pragma unroll
    for (int e = 0; e < TB_E; ++e) g[e] *= inv;
    int i1 = 0;
    #pragma unroll
    for (int e = 1; e < TB_E; ++e) if (g[e] > g[i1]) i1 = e;
    int i2 = (i1 == 0) ? 1 : 0;
    #pragma unroll
    for (int e = 0; e < TB_E; ++e) if (e != i1 && g[e] > g[i2]) i2 = e;
    #pragma unroll
    for (int e = 0; e < TB_E; ++e)
        ew[(long)t * TB_E + e] = (e == i1) ? g[i1] : ((e == i2) ? g[i2] : 0.f);
}

__global__ __launch_bounds__(256)
void zero4_kernel(float4* __restrict__ p) {
    long i = (long)blockIdx.x * blockDim.x + threadIdx.x;
    p[i] = make_float4(0.f, 0.f, 0.f, 0.f);
}

__global__ __launch_bounds__(256)
void add3_kernel(const float4* __restrict__ a, const float4* __restrict__ b,
                 const float4* __restrict__ c, float4* __restrict__ y) {
    long i = (long)blockIdx.x * blockDim.x + threadIdx.x;
    float4 va = a[i], vb = b[i], vc = c[i];
    y[i] = make_float4(va.x + vb.x + vc.x, va.y + vb.y + vc.y,
                       va.z + vb.z + vc.z, va.w + vb.w + vc.w);
}

// ---------------------------------------------------------------------------
extern "C" void kernel_launch(void* const* d_in, const int* in_sizes, int n_in,
                              void* d_out, int out_size, void* d_ws, size_t ws_size,
                              hipStream_t stream) {
    (void)in_sizes; (void)n_in; (void)out_size; (void)ws_size;

    const int*   tokens = (const int*)d_in[0];
    const float* embed  = (const float*)d_in[1];
    const float* ln1_s  = (const float*)d_in[2];
    const float* ln1_b  = (const float*)d_in[3];
    const float* qkv_w  = (const float*)d_in[4];
    const float* qkv_b  = (const float*)d_in[5];
    const float* gate_w = (const float*)d_in[6];
    const float* gate_b = (const float*)d_in[7];
    const float* exp_w  = (const float*)d_in[8];
    const float* exp_b  = (const float*)d_in[9];
    const float* ln2_s  = (const float*)d_in[10];
    const float* ln2_b  = (const float*)d_in[11];
    const float* ffn1_w = (const float*)d_in[12];
    const float* ffn1_b = (const float*)d_in[13];
    const float* ffn2_w = (const float*)d_in[14];
    const float* ffn2_b = (const float*)d_in[15];
    const float* out_w  = (const float*)d_in[16];
    const float* out_b  = (const float*)d_in[17];
    float* out = (float*)d_out;

    const long S = TB_S, D = TB_D, H = TB_H, DH = TB_DH, E = TB_E,
               DFF = TB_DFF, V = TB_V;

    float* ws = (float*)d_ws;
    size_t off = 0;
    auto alloc = [&](size_t n) { float* p = ws + off; off += n; return p; };
    float* xbuf   = alloc(S * D);
    float* xln    = alloc(S * D);
    float* qkvbuf = alloc(S * 3 * D);
    float* scores = alloc(H * S * S);
    float* attnb  = alloc(S * D);
    float* gateb  = alloc(S * E);
    float* ewb    = alloc(S * E);
    float* moeb   = alloc(S * D);
    float* ffb    = alloc(S * DFF);

    embed_kernel<<<TB_S, 256, 0, stream>>>(tokens, embed, xbuf);

    for (int l = 0; l < TB_L; ++l) {
        ln_kernel<<<TB_S, 256, 0, stream>>>(xbuf, ln1_s + l * D, ln1_b + l * D, xln);

        // qkv = xln @ qkv_w[l] + qkv_b[l]   (1024 x 2048 x 6144)
        {
            dim3 g(3 * TB_D / 64, TB_S / 128, 1);
            wmma_gemm<<<g, 128, 0, stream>>>(
                xln, D, 0,
                qkv_w + (long)l * D * 3 * D, 3 * D, 1, 0,
                qkv_b + (long)l * 3 * D, 0,
                qkvbuf, 3 * D, 0,
                nullptr, 0, 0,
                TB_S, 3 * TB_D, TB_D, 1.0f, 0, 0);
        }
        // scores[h] = (Q_h @ K_h^T) / sqrt(DH)    per head, z = H
        {
            dim3 g(TB_S / 64, TB_S / 128, TB_H);
            wmma_gemm<<<g, 128, 0, stream>>>(
                qkvbuf, 3 * D, DH,            // A = Q head (strided rows)
                qkvbuf + D, 1, 3 * D, DH,     // W[k][n] = K[n][k] (transposed)
                nullptr, 0,
                scores, S, S * S,
                nullptr, 0, 0,
                TB_S, TB_S, TB_DH, 0.08838834764831845f, 0, 0);
        }
        softmax1024<<<TB_H * TB_S, 256, 0, stream>>>(scores);
        // attn_out[h] = attn_h @ V_h   (1024 x 1024 x 128), z = H
        {
            dim3 g(TB_DH / 64, TB_S / 128, TB_H);
            wmma_gemm<<<g, 128, 0, stream>>>(
                scores, S, S * S,
                qkvbuf + 2 * D, 3 * D, 1, DH,
                nullptr, 0,
                attnb, D, DH,
                nullptr, 0, 0,
                TB_S, TB_DH, TB_S, 1.0f, 0, 0);
        }
        // gating
        gate_kernel<<<TB_S, 256, 0, stream>>>(xln, gate_w + (long)l * D * E,
                                              gate_b + (long)l * E, gateb);
        moe_topk_kernel<<<(TB_S + 255) / 256, 256, 0, stream>>>(gateb, ewb);
        // moe_out = sum_e ew[:,e] * (xln @ exp_w[l,e] + exp_b[l,e])  (atomic accum)
        zero4_kernel<<<(TB_S * TB_D / 4) / 256, 256, 0, stream>>>((float4*)moeb);
        {
            dim3 g(TB_D / 64, TB_S / 128, TB_E);
            wmma_gemm<<<g, 128, 0, stream>>>(
                xln, D, 0,
                exp_w + (long)l * E * D * D, D, 1, D * D,
                exp_b + (long)l * E * D, D,
                moeb, D, 0,
                ewb, E, 1,
                TB_S, TB_D, TB_D, 1.0f, 0, 1);
        }
        // x = residual(=xln) + attn + moe
        add3_kernel<<<(TB_S * TB_D / 4) / 256, 256, 0, stream>>>(
            (const float4*)xln, (const float4*)attnb, (const float4*)moeb,
            (float4*)xbuf);

        ln_kernel<<<TB_S, 256, 0, stream>>>(xbuf, ln2_s + l * D, ln2_b + l * D, xln);

        // ffn1 + gelu  (1024 x 2048 x 8192)
        {
            dim3 g(TB_DFF / 64, TB_S / 128, 1);
            wmma_gemm<<<g, 128, 0, stream>>>(
                xln, D, 0,
                ffn1_w + (long)l * D * DFF, DFF, 1, 0,
                ffn1_b + (long)l * DFF, 0,
                ffb, DFF, 0,
                nullptr, 0, 0,
                TB_S, TB_DFF, TB_D, 1.0f, 1, 0);
        }
        // ffn2  (1024 x 8192 x 2048)
        {
            dim3 g(TB_D / 64, TB_S / 128, 1);
            wmma_gemm<<<g, 128, 0, stream>>>(
                ffb, DFF, 0,
                ffn2_w + (long)l * DFF * D, D, 1, 0,
                ffn2_b + (long)l * D, 0,
                xbuf, D, 0,
                nullptr, 0, 0,
                TB_S, TB_D, TB_DFF, 1.0f, 0, 0);
        }
    }

    // logits = x @ out_w + out_b   (1024 x 2048 x 32000)
    {
        dim3 g(TB_V / 64, TB_S / 128, 1);
        wmma_gemm<<<g, 128, 0, stream>>>(
            xbuf, D, 0,
            out_w, V, 1, 0,
            out_b, 0,
            out, V, 0,
            nullptr, 0, 0,
            TB_S, TB_V, TB_D, 1.0f, 0, 0);
    }
}
